// MVIGenerator_14791867367730
// MI455X (gfx1250) — compile-verified, matrix operand order checked by reading, not run on hardware
//
#include <hip/hip_runtime.h>
#include <hip/hip_bf16.h>
#include <stdint.h>

// ---------------------------------------------------------------------------
// Contextual-attention pipeline for MI455X (gfx1250, wave32, WMMA).
// B=8, C=128, H=W=64; downsampled 32x32 grids; L=Q=1024; K=C*9=1152.
// Heavy work: scores GEMM (1024x1024x1152) and 16 deconv GEMMs
// (128x1024x1024) per batch, via v_wmma_f32_16x16x32_bf16 with
// double-buffered GLOBAL_LOAD_ASYNC_TO_LDS_B128 tile staging (ASYNCcnt).
// ---------------------------------------------------------------------------

typedef __attribute__((ext_vector_type(16))) __bf16 bf16x16;
typedef __attribute__((ext_vector_type(8)))  __bf16 bf16x8;
typedef __attribute__((ext_vector_type(8)))  float  f32x8;

#define CDIM   128
#define LP     1024          // background patches == query locations
#define KDIM   1152          // C * 3 * 3
#define SCALE_F 10.0f
#define EPS_F   0.0001f

// LDS tile geometry: 128 rows x 32 K-elems, row stride 48 elems (96B, 16B-aligned)
#define ROWSTR   48
#define TILE_ELE (128 * ROWSTR)         // 6144 ushorts per operand tile
#define BUF_BYTE (2 * TILE_ELE * 2)     // 24576 B per double-buffer stage

static __device__ __forceinline__ unsigned short f2bf(float x) {
  union { float f; unsigned u; } v; v.f = x;
  unsigned r = v.u + 0x7FFFu + ((v.u >> 16) & 1u);   // round-to-nearest-even
  return (unsigned short)(r >> 16);
}

// ---------------------------------------------------------------------------
// mask -> mm[l] : 1.0 where the 3x3 patch of the /8-downsampled mask is all 0
// ---------------------------------------------------------------------------
__global__ void mm_kernel(const float* __restrict__ mask, float* __restrict__ mm) {
  int l = blockIdx.x * blockDim.x + threadIdx.x;
  if (l >= LP) return;
  int i = l >> 5, j = l & 31;
  float s = 0.0f;
  for (int di = -1; di <= 1; ++di)
    for (int dj = -1; dj <= 1; ++dj) {
      int y = i + di, x = j + dj;
      if ((unsigned)y < 32u && (unsigned)x < 32u)
        s += mask[(8 * y) * 256 + 8 * x];            // batch 0, channel 0
    }
  mm[l] = (s == 0.0f) ? 1.0f : 0.0f;
}

// ---------------------------------------------------------------------------
// invnorm[l] = 1 / max(||w_patch_l||, EPS) over bd = b[:, ::2, ::2]
// ---------------------------------------------------------------------------
__global__ __launch_bounds__(256)
void invnorm_kernel(const float* __restrict__ b, float* __restrict__ invn) {
  __shared__ float red[256];
  int l = blockIdx.x;
  int i = l >> 5, j = l & 31;
  float s = 0.0f;
  for (int idx = threadIdx.x; idx < KDIM; idx += 256) {
    int c = idx / 9, tap = idx - 9 * c;
    int y = i + tap / 3 - 1, x = j + tap % 3 - 1;
    if ((unsigned)y < 32u && (unsigned)x < 32u) {
      float v = b[(c * 64 + 2 * y) * 64 + 2 * x];
      s += v * v;
    }
  }
  red[threadIdx.x] = s;
  __syncthreads();
  for (int off = 128; off > 0; off >>= 1) {
    if (threadIdx.x < off) red[threadIdx.x] += red[threadIdx.x + off];
    __syncthreads();
  }
  if (threadIdx.x == 0) invn[l] = 1.0f / fmaxf(sqrtf(red[0]), EPS_F);
}

// ---------------------------------------------------------------------------
// Materialize bf16 GEMM operands (row x K, K contiguous):
//   Aq[q][k] = query patch values from f[::2,::2]
//   Bw[l][k] = normalized background patch values from b[::2,::2]
// ---------------------------------------------------------------------------
__global__ void build_ab_kernel(const float* __restrict__ f, const float* __restrict__ b,
                                const float* __restrict__ invn,
                                unsigned short* __restrict__ Aq,
                                unsigned short* __restrict__ Bw) {
  int idx = blockIdx.x * blockDim.x + threadIdx.x;
  if (idx >= LP * KDIM) return;
  int r = idx / KDIM, k = idx - r * KDIM;
  int c = k / 9, tap = k - 9 * c;
  int i = r >> 5, j = r & 31;
  int y = i + tap / 3 - 1, x = j + tap % 3 - 1;
  float fv = 0.0f, bv = 0.0f;
  if ((unsigned)y < 32u && (unsigned)x < 32u) {
    int o = (c * 64 + 2 * y) * 64 + 2 * x;
    fv = f[o];
    bv = b[o];
  }
  Aq[idx] = f2bf(fv);
  Bw[idx] = f2bf(bv * invn[r]);
}

// ---------------------------------------------------------------------------
// Wd[kpos][c][l] = raw 4x4 patch value b[c, 2i+kh-1, 2j+kw-1]  (zero padded)
// ---------------------------------------------------------------------------
__global__ void wd_kernel(const float* __restrict__ b, unsigned short* __restrict__ Wd) {
  int idx = blockIdx.x * blockDim.x + threadIdx.x;
  if (idx >= 16 * CDIM * LP) return;
  int kpos = idx / (CDIM * LP);
  int rem  = idx - kpos * (CDIM * LP);
  int c = rem / LP, l = rem - c * LP;
  int kh = kpos >> 2, kw = kpos & 3;
  int i = l >> 5, j = l & 31;
  int y = 2 * i + kh - 1, x = 2 * j + kw - 1;
  float v = 0.0f;
  if ((unsigned)y < 64u && (unsigned)x < 64u) v = b[(c * 64 + y) * 64 + x];
  Wd[idx] = f2bf(v);
}

// ---------------------------------------------------------------------------
// WMMA GEMM: C[m][n] = sum_k A[m][k] * B[n][k]   (both row x K, bf16, K%32==0)
// Block tile 128(M) x 128(N): 8 waves as 4(M) x 2(N), wave tile 32x64 = 2x4
// 16x16 accumulators. K staged 32-wide via async global->LDS b128 loads,
// double buffered (4 async loads / thread / step; s_wait_asynccnt 0x4
// exploits in-order ASYNCcnt completion to overlap prefetch with WMMA).
// ---------------------------------------------------------------------------
template <int KTOT, bool DECONV>
__global__ __launch_bounds__(256)
void wmma_gemm(const unsigned short* __restrict__ Amat,
               const unsigned short* __restrict__ Bmat,
               float* __restrict__ Cout,   // scores [L][Q]   (!DECONV)
               float* __restrict__ Yout) { // y batch [128][64][64] (DECONV)
  __shared__ __align__(16) unsigned short lds[2 * 2 * TILE_ELE];

  const int tid  = threadIdx.x;
  const int lane = tid & 31;
  const int wave = tid >> 5;
  const int wm   = wave & 3;       // M sub-block 0..3  (32 rows each)
  const int wn   = wave >> 2;      // N sub-block 0..1  (64 cols each)
  const int lrow = lane & 15;
  const int half = lane >> 4;

  const int M0 = blockIdx.y * 128;
  const int N0 = blockIdx.x * 128;
  const unsigned short* Ab = Amat + (DECONV ? (size_t)blockIdx.z * CDIM * LP : 0);

  // LDS byte offset of this workgroup's tile (generic ptr low 32 bits == LDS addr)
  const unsigned ldsbase = (unsigned)(size_t)&lds[0];

  // staging chunk coordinates: 512 chunks of 16B per 128x32 tile, 2 per thread
  const int r0 = tid >> 2;                 // rows 0..63
  const int c0 = (tid & 3) * 8;            // K element 0/8/16/24
  const int r1 = r0 + 64;                  // rows 64..127

  auto issue = [&](int k0, unsigned bufofs) {
    unsigned la, ga;
    // A tile chunks
    la = ldsbase + bufofs + (unsigned)(r0 * ROWSTR + c0) * 2u;
    ga = (unsigned)(((size_t)(M0 + r0) * KTOT + k0 + c0) * 2);
    asm volatile("global_load_async_to_lds_b128 %0, %1, %2"
                 :: "v"(la), "v"(ga), "s"(Ab) : "memory");
    la = ldsbase + bufofs + (unsigned)(r1 * ROWSTR + c0) * 2u;
    ga = (unsigned)(((size_t)(M0 + r1) * KTOT + k0 + c0) * 2);
    asm volatile("global_load_async_to_lds_b128 %0, %1, %2"
                 :: "v"(la), "v"(ga), "s"(Ab) : "memory");
    // B tile chunks (second half of the buffer)
    la = ldsbase + bufofs + (unsigned)(TILE_ELE + r0 * ROWSTR + c0) * 2u;
    ga = (unsigned)(((size_t)(N0 + r0) * KTOT + k0 + c0) * 2);
    asm volatile("global_load_async_to_lds_b128 %0, %1, %2"
                 :: "v"(la), "v"(ga), "s"(Bmat) : "memory");
    la = ldsbase + bufofs + (unsigned)(TILE_ELE + r1 * ROWSTR + c0) * 2u;
    ga = (unsigned)(((size_t)(N0 + r1) * KTOT + k0 + c0) * 2);
    asm volatile("global_load_async_to_lds_b128 %0, %1, %2"
                 :: "v"(la), "v"(ga), "s"(Bmat) : "memory");
  };

  const f32x8 zero8 = {0.f, 0.f, 0.f, 0.f, 0.f, 0.f, 0.f, 0.f};
  f32x8 acc[2][4] = {{zero8, zero8, zero8, zero8}, {zero8, zero8, zero8, zero8}};

  issue(0, 0u);
  int buf = 0;
  for (int k0 = 0; k0 < KTOT; k0 += 32) {
    if (k0 + 32 < KTOT) {
      issue(k0 + 32, (buf ^ 1) ? (unsigned)BUF_BYTE : 0u);
      asm volatile("s_wait_asynccnt 0x4" ::: "memory");  // current buffer landed
    } else {
      asm volatile("s_wait_asynccnt 0x0" ::: "memory");
    }
    __syncthreads();

    const unsigned short* As = &lds[buf * 2 * TILE_ELE];
    const unsigned short* Bs = As + TILE_ELE;

    // ISA 7.12.2 16-bit operand layout: e<8 -> K=8*half+e ; e>=8 -> K=16+8*half+(e-8)
    bf16x16 afr[2], bfr[4];
#pragma unroll
    for (int tm = 0; tm < 2; ++tm) {
      int r = wm * 32 + tm * 16 + lrow;
      bf16x8 lo = *reinterpret_cast<const bf16x8*>(&As[r * ROWSTR + half * 8]);
      bf16x8 hi = *reinterpret_cast<const bf16x8*>(&As[r * ROWSTR + 16 + half * 8]);
      afr[tm] = __builtin_shufflevector(lo, hi, 0,1,2,3,4,5,6,7,8,9,10,11,12,13,14,15);
    }
#pragma unroll
    for (int tn = 0; tn < 4; ++tn) {
      int r = wn * 64 + tn * 16 + lrow;
      bf16x8 lo = *reinterpret_cast<const bf16x8*>(&Bs[r * ROWSTR + half * 8]);
      bf16x8 hi = *reinterpret_cast<const bf16x8*>(&Bs[r * ROWSTR + 16 + half * 8]);
      bfr[tn] = __builtin_shufflevector(lo, hi, 0,1,2,3,4,5,6,7,8,9,10,11,12,13,14,15);
    }
#pragma unroll
    for (int tm = 0; tm < 2; ++tm)
#pragma unroll
      for (int tn = 0; tn < 4; ++tn)
        acc[tm][tn] = __builtin_amdgcn_wmma_f32_16x16x32_bf16(
            false, afr[tm], false, bfr[tn], (short)0, acc[tm][tn], false, false);
    __syncthreads();   // protect this buffer from next iteration's prefetch
    buf ^= 1;
  }

  // D layout: VGPR v -> m = v + 8*half ; n = lane%16
#pragma unroll
  for (int tm = 0; tm < 2; ++tm)
#pragma unroll
    for (int tn = 0; tn < 4; ++tn) {
      int mbase = M0 + wm * 32 + tm * 16 + half * 8;
      int nidx  = N0 + wn * 64 + tn * 16 + lrow;
#pragma unroll
      for (int v = 0; v < 8; ++v) {
        float val = acc[tm][tn][v];
        int m = mbase + v;
        if (!DECONV) {
          Cout[(size_t)m * 1024 + nidx] = val;
        } else {
          int kh = (int)blockIdx.z >> 2, kw = (int)blockIdx.z & 3;
          int i = nidx >> 5, j = nidx & 31;
          int th = 2 * i + kh - 1, tw = 2 * j + kw - 1;
          if ((unsigned)th < 64u && (unsigned)tw < 64u)
            unsafeAtomicAdd(&Yout[((size_t)m * 64 + th) * 64 + tw], 0.25f * val);
        }
      }
    }
}

// ---------------------------------------------------------------------------
// Masked softmax over l for each query q; zeros (masked) contribute exp(0).
// Writes yi transposed as bf16 [q][l] so deconv GEMM is K-contiguous.
// ---------------------------------------------------------------------------
__global__ __launch_bounds__(256)
void softmax_kernel(const float* __restrict__ scores, const float* __restrict__ mm,
                    unsigned short* __restrict__ yib) {
  __shared__ float red[4][64];
  int tid = threadIdx.x;
  int tq = tid & 63, tl = tid >> 6;
  int q = blockIdx.x * 64 + tq;

  float mx = -1e30f;
  for (int l = tl; l < LP; l += 4) {
    float x = scores[(size_t)l * 1024 + q] * mm[l] * SCALE_F;
    mx = fmaxf(mx, x);
  }
  red[tl][tq] = mx;
  __syncthreads();
  if (tl == 0)
    red[0][tq] = fmaxf(fmaxf(red[0][tq], red[1][tq]), fmaxf(red[2][tq], red[3][tq]));
  __syncthreads();
  mx = red[0][tq];
  __syncthreads();

  float sum = 0.0f;
  for (int l = tl; l < LP; l += 4) {
    float x = scores[(size_t)l * 1024 + q] * mm[l] * SCALE_F;
    sum += __expf(x - mx);
  }
  red[tl][tq] = sum;
  __syncthreads();
  if (tl == 0)
    red[0][tq] = red[0][tq] + red[1][tq] + red[2][tq] + red[3][tq];
  __syncthreads();
  float inv = 1.0f / red[0][tq];

  for (int l = tl; l < LP; l += 4) {
    float x = scores[(size_t)l * 1024 + q] * mm[l] * SCALE_F;
    float p = __expf(x - mx) * inv * mm[l];
    yib[(size_t)q * 1024 + l] = f2bf(p);
  }
}

// ---------------------------------------------------------------------------
extern "C" void kernel_launch(void* const* d_in, const int* in_sizes, int n_in,
                              void* d_out, int out_size, void* d_ws, size_t ws_size,
                              hipStream_t stream) {
  const float* f    = (const float*)d_in[0];
  const float* b    = (const float*)d_in[1];
  const float* mask = (const float*)d_in[2];
  float* y = (float*)d_out;

  // workspace carve-up (~15.2 MB total, reused across the 8 batches)
  char* ws = (char*)d_ws;
  size_t off = 0;
  auto carve = [&](size_t bytes) -> void* {
    void* p = ws + off;
    off = (off + bytes + 255) & ~(size_t)255;
    return p;
  };
  float*          mm     = (float*)carve(LP * sizeof(float));
  float*          invn   = (float*)carve(LP * sizeof(float));
  unsigned short* Aq     = (unsigned short*)carve((size_t)LP * KDIM * 2);
  unsigned short* Bw     = (unsigned short*)carve((size_t)LP * KDIM * 2);
  float*          scores = (float*)carve((size_t)LP * 1024 * sizeof(float));
  unsigned short* yib    = (unsigned short*)carve((size_t)1024 * LP * 2);
  unsigned short* Wd     = (unsigned short*)carve((size_t)16 * CDIM * LP * 2);
  (void)ws_size; (void)in_sizes; (void)n_in;

  hipMemsetAsync(d_out, 0, (size_t)out_size * sizeof(float), stream);
  mm_kernel<<<4, 256, 0, stream>>>(mask, mm);

  for (int bat = 0; bat < 8; ++bat) {
    const float* fb = f + (size_t)bat * CDIM * 64 * 64;
    const float* bb = b + (size_t)bat * CDIM * 64 * 64;
    float*       yb = y + (size_t)bat * CDIM * 64 * 64;

    invnorm_kernel<<<LP, 256, 0, stream>>>(bb, invn);
    build_ab_kernel<<<(LP * KDIM + 255) / 256, 256, 0, stream>>>(fb, bb, invn, Aq, Bw);
    // scores[l][q] : A rows = Bw (M=L), B rows = Aq (N=Q)
    wmma_gemm<KDIM, false><<<dim3(8, 8, 1), 256, 0, stream>>>(Bw, Aq, scores, nullptr);
    softmax_kernel<<<16, 256, 0, stream>>>(scores, mm, yib);
    wd_kernel<<<(16 * CDIM * LP + 255) / 256, 256, 0, stream>>>(bb, Wd);
    // 16 deconv slices: A rows = Wd[kpos] (M=C=128), B rows = yib (N=S=1024)
    wmma_gemm<LP, true><<<dim3(8, 1, 16), 256, 0, stream>>>(Wd, yib, nullptr, yb);
  }
}
